// PromptGenerate_73203422593166
// MI455X (gfx1250) — matrix operand order repaired and checked by reading, not compiled, not run on hardware
//
#include <hip/hip_runtime.h>
#include <hip/hip_bf16.h>
#include <math.h>

// ---------------------------------------------------------------------------
// Saliency prompt generation for MI455X (gfx1250, wave32).
// conv1 (12ch 3x3 -> 32ch) runs as an implicit GEMM on the WMMA pipe:
//   D(16 px x 16 ch) += A(16 x 32 patch-slice, f16) x B(32 x 16 w, f16)
// A is staged in LDS (im2col tile) and fragments are loaded as two 16-byte
// ds_load_b128s per the ISA VGPR layout; B is pre-swizzled to fragment order
// so each lane does one contiguous 32B global load. 4 K-steps x 2 N-tiles =
// 8 v_wmma per wave, fused with bias+BN+GELU+1x1 conv+sigmoid.
// Fill loads are unconditional with clamped addresses (select-zero after) so
// they form one vmem clause instead of 16 exec-predicated singles.
// ---------------------------------------------------------------------------

typedef _Float16 h16;
typedef __attribute__((ext_vector_type(16))) _Float16 v16h;
typedef __attribute__((ext_vector_type(8)))  _Float16 v8h;
typedef __attribute__((ext_vector_type(8)))  float    v8f;

#define BATCH 8
#define HS 512
#define HW (HS*HS)            // 262144 saliency pixels per image
#define NPTS 10
#define KPEAKS 512
#define CANDS 4096

// workspace layout (bytes); requires ~59 MB
#define WS_FREQ  0ull                                   // B*12*HW halves
#define WS_WB    (WS_FREQ + (size_t)BATCH*12*HW*2)      // 4*2*32*16 halves
#define WS_SAL   (WS_WB + 4096*2)                       // B*HW floats
#define WS_BOXI  (WS_SAL + (size_t)BATCH*HW*4)          // B*4 ints

// output layout (floats): coords | labels | boxes | masks
#define OUT_LABELS 160
#define OUT_BOXES  240
#define OUT_MASKS  272

// ---------------- stage 1: gray + Haar DWT + edge energy (fp16 out) --------
__global__ void __launch_bounds__(256)
k_freq(const float* __restrict__ rgb, const float* __restrict__ dep,
       h16* __restrict__ fq)
{
  int t = blockIdx.x*256 + threadIdx.x;
  if (t >= BATCH*HW) return;
  int b = t >> 18;
  int pos = t & (HW-1);
  int u = pos >> 9, v = pos & 511;
  int y0 = 2*u, x0 = 2*v;
  size_t rb = (size_t)b*3*1048576;
  size_t db = (size_t)b*1048576;
#define GRAYV(yy,xx) ((rgb[rb + (size_t)(yy)*1024 + (xx)] + \
                       rgb[rb + 1048576 + (size_t)(yy)*1024 + (xx)] + \
                       rgb[rb + 2097152 + (size_t)(yy)*1024 + (xx)]) * (1.f/3.f))
  float g00 = GRAYV(y0,   x0), g10 = GRAYV(y0+1, x0);
  float g01 = GRAYV(y0,   x0+1), g11 = GRAYV(y0+1, x0+1);
#undef GRAYV
  float d00 = dep[db + (size_t)y0*1024 + x0];
  float d10 = dep[db + (size_t)(y0+1)*1024 + x0];
  float d01 = dep[db + (size_t)y0*1024 + x0+1];
  float d11 = dep[db + (size_t)(y0+1)*1024 + x0+1];
  float rLL = 0.5f*(g00+g10), rLH = 0.5f*(g00-g10);
  float rHL = 0.5f*(g01+g11), rHH = 0.5f*(g01-g11);
  float dLL = 0.5f*(d00+d10), dLH = 0.5f*(d00-d10);
  float dHL = 0.5f*(d01+d11), dHH = 0.5f*(d01-d11);
  float Er = sqrtf(rLH*rLH + rHL*rHL + rHH*rHH + 1e-8f);
  float Ed = sqrtf(dLH*dLH + dHL*dHL + dHH*dHH + 1e-8f);
  float vals[10] = {rLL, rLH, rHL, rHH, dLL, dLH, dHL, dHH, Er, Ed};
  for (int c = 0; c < 10; ++c)
    fq[((size_t)(b*12 + c) << 18) + pos] = (h16)vals[c];
}

// ---------------- stage 2: 5x5 gaussian smooth of energy channels ----------
__global__ void __launch_bounds__(256)
k_smooth(h16* __restrict__ fq)
{
  int t = blockIdx.x*256 + threadIdx.x;
  if (t >= BATCH*2*HW) return;
  int b = t / (2*HW);
  int rem = t - b*2*HW;
  int ch = rem >> 18;
  int pos = rem & (HW-1);
  int u = pos >> 9, v = pos & 511;
  const float w1[5] = {0.13533528f, 0.60653066f, 1.f, 0.60653066f, 0.13533528f};
  const float S1 = 0.13533528f*2.f + 0.60653066f*2.f + 1.f;
  const float inv = 1.f/(S1*S1);
  const h16* src = fq + ((size_t)(b*12 + 8 + ch) << 18);
  float acc = 0.f;
  for (int i = 0; i < 5; ++i) {
    int yy = u + i - 2;
    if ((unsigned)yy >= 512u) continue;
    for (int j = 0; j < 5; ++j) {
      int xx = v + j - 2;
      if ((unsigned)xx >= 512u) continue;
      acc += w1[i]*w1[j]*(float)src[(yy<<9)+xx];
    }
  }
  fq[((size_t)(b*12 + 10 + ch) << 18) + pos] = (h16)(acc*inv);
}

// ---------------- weight swizzle: (32,12,3,3) -> WMMA B-fragment order -----
// wB[((s*2 + t)*32 + lane)*16 + h] = w1[n][c][ky][kx] as f16, where
// k = s*32 + (h>>3)*16 + (lane>>4)*8 + (h&7) (zero-padded past 108), n = t*16 + (lane&15).
__global__ void __launch_bounds__(256)
k_wt(const float* __restrict__ w, h16* __restrict__ wB)
{
  int t = blockIdx.x*256 + threadIdx.x;
  if (t >= 4096) return;
  int h = t & 15;
  int lane = (t >> 4) & 31;
  int nt = (t >> 9) & 1;
  int s = t >> 10;
  int k = s*32 + (h >> 3)*16 + (lane >> 4)*8 + (h & 7);
  int n = nt*16 + (lane & 15);
  float val = 0.f;
  if (k < 108) {
    int c = k / 9, j = k - c*9;
    int ky = j / 3, kx = j - ky*3;
    val = w[((n*12 + c)*3 + ky)*3 + kx];
  }
  wB[t] = (h16)val;
}

// ---------------- fused conv1(WMMA)+BN+GELU+conv2+sigmoid ------------------
// grid = BATCH*2048 blocks, 256 threads = 8 waves, 16 pixels per wave.
__global__ void __launch_bounds__(256)
k_conv(const h16* __restrict__ fq, const h16* __restrict__ wB,
       const float* __restrict__ cb1, const float* __restrict__ gamma,
       const float* __restrict__ beta, const float* __restrict__ mean,
       const float* __restrict__ var, const float* __restrict__ w2,
       const float* __restrict__ cb2, float* __restrict__ sal)
{
  // im2col tile: 8 waves x 16 pixels x 32 k-values of f16
  __shared__ __align__(16) h16 lds_a[8*512];
  int tid = threadIdx.x;
  int wave = tid >> 5, lane = tid & 31;
  int b = blockIdx.x >> 11;
  int blk = blockIdx.x & 2047;
  int base = blk*128 + wave*16;
  int m = lane & 15, g = lane >> 4;
  const h16* fqb = fq + ((size_t)(b*12) << 18);
  h16* atile = lds_a + wave*512;
  // base is 16-aligned -> all 16 pixels of this wave share one source row
  int py = base >> 9, px0 = base & 511;
  v8f acc0 = {}, acc1 = {};
#pragma unroll
  for (int s = 0; s < 4; ++s) {
    // ---- fill: lane owns k-column (k = lane); decode (c,dy,dx) once ----
    int kg = s*32 + lane;
    bool ok = kg < 108;
    int c = 0, dy = 0, dx = 0;
    if (ok) {
      c = kg / 9;
      int j = kg - c*9;
      int jy = j / 3;
      dy = jy - 1;
      dx = j - jy*3 - 1;
    }
    int yy = py + dy;
    bool rowok = ok && (unsigned)yy < 512u;
    int yc = min(max(yy, 0), 511);
    const h16* row = fqb + ((size_t)c << 18) + (yc << 9);
    int x0 = px0 + dx;
    // unconditional clamped loads (clause-friendly), select-zero after
    h16 vals[16];
#pragma unroll
    for (int i = 0; i < 16; ++i) {
      int xx = x0 + i;
      int xc = min(max(xx, 0), 511);
      h16 vv = row[xc];
      vals[i] = (rowok && (unsigned)xx < 512u) ? vv : (h16)0.f;
    }
#pragma unroll
    for (int i = 0; i < 16; ++i)
      atile[i*32 + lane] = vals[i];
    __syncthreads();
    // ---- A fragment: lane = row m; two contiguous 8-half runs (ISA map:
    //      k_local = g*8 + [0..7] and 16 + g*8 + [0..7]) -> 2x ds_load_b128
    union { v16h v; v8h h[2]; } ua;
    ua.h[0] = *(const v8h*)&atile[m*32 + g*8];
    ua.h[1] = *(const v8h*)&atile[m*32 + g*8 + 16];
    // ---- B fragments: pre-swizzled, one contiguous 32B run per lane ----
    v16h bf0 = *(const v16h*)&wB[((s*2 + 0)*32 + lane)*16];
    v16h bf1 = *(const v16h*)&wB[((s*2 + 1)*32 + lane)*16];
    acc0 = __builtin_amdgcn_wmma_f32_16x16x32_f16(false, ua.v, false, bf0,
                                                  (short)0, acc0, false, false);
    acc1 = __builtin_amdgcn_wmma_f32_16x16x32_f16(false, ua.v, false, bf1,
                                                  (short)0, acc1, false, false);
    __syncthreads();
  }
  // D layout: lane holds channel n = lane&15 (+16 for acc1), row m = r + 8*g
  int n0 = lane & 15, n1 = 16 + n0;
  float bi0 = cb1[n0], bi1 = cb1[n1];
  float ga0 = gamma[n0], ga1 = gamma[n1];
  float be0 = beta[n0], be1 = beta[n1];
  float mu0 = mean[n0], mu1 = mean[n1];
  float iv0 = rsqrtf(var[n0] + 1e-5f), iv1 = rsqrtf(var[n1] + 1e-5f);
  float wo0 = w2[n0], wo1 = w2[n1];
  float bias2 = cb2[0];
  float* salb = sal + ((size_t)b << 18);
  const float kG = 0.7978845608028654f; // sqrt(2/pi), tanh-gelu
#pragma unroll
  for (int r = 0; r < 8; ++r) {
    float x0f = acc0[r] + bi0;
    float x1f = acc1[r] + bi1;
    x0f = ga0*(x0f - mu0)*iv0 + be0;
    x1f = ga1*(x1f - mu1)*iv1 + be1;
    float e0 = 0.5f*x0f*(1.f + tanhf(kG*(x0f + 0.044715f*x0f*x0f*x0f)));
    float e1 = 0.5f*x1f*(1.f + tanhf(kG*(x1f + 0.044715f*x1f*x1f*x1f)));
    float sum = e0*wo0 + e1*wo1;
    // reduce over 16 lanes of each wave32 half (channels 0..31)
    sum += __shfl_xor(sum, 1, 32);
    sum += __shfl_xor(sum, 2, 32);
    sum += __shfl_xor(sum, 4, 32);
    sum += __shfl_xor(sum, 8, 32);
    if (n0 == 0) {
      int pix = base + r + 8*g;
      salb[pix] = 1.f/(1.f + __expf(-(sum + bias2)));
    }
  }
}

// ---------------- bilinear upsample 512->1024 (half-pixel centers) ---------
__global__ void __launch_bounds__(256)
k_resize(const float* __restrict__ sal, float* __restrict__ masks)
{
  int t = blockIdx.x*256 + threadIdx.x;
  if (t >= BATCH*1048576) return;
  int b = t >> 20;
  int pos = t & 1048575;
  int oy = pos >> 10, ox = pos & 1023;
  float sy = fminf(fmaxf(oy*0.5f - 0.25f, 0.f), 511.f);
  float sx = fminf(fmaxf(ox*0.5f - 0.25f, 0.f), 511.f);
  int y0 = (int)sy, x0 = (int)sx;
  int y1 = min(y0+1, 511), x1 = min(x0+1, 511);
  float fy = sy - y0, fx = sx - x0;
  const float* sb = sal + ((size_t)b << 18);
  float v00 = sb[(y0<<9)+x0], v01 = sb[(y0<<9)+x1];
  float v10 = sb[(y1<<9)+x0], v11 = sb[(y1<<9)+x1];
  masks[t] = (v00*(1.f-fx)+v01*fx)*(1.f-fy) + (v10*(1.f-fx)+v11*fx)*fy;
}

// ---------------- bbox from thresholded masks ------------------------------
__global__ void k_boxinit(int* boxi)
{
  int t = threadIdx.x;
  if (t < BATCH) {
    boxi[t*4+0] = 1024;  // rmin
    boxi[t*4+1] = -1;    // rmax
    boxi[t*4+2] = 1024;  // cmin
    boxi[t*4+3] = -1;    // cmax
  }
}

__global__ void __launch_bounds__(256)
k_boxreduce(const float* __restrict__ masks, int* __restrict__ boxi)
{
  int t = blockIdx.x*256 + threadIdx.x;
  if (t >= BATCH*1048576) return;
  if (masks[t] > 0.5f) {
    int b = t >> 20;
    int pos = t & 1048575;
    int y = pos >> 10, x = pos & 1023;
    atomicMin(&boxi[b*4+0], y);
    atomicMax(&boxi[b*4+1], y);
    atomicMin(&boxi[b*4+2], x);
    atomicMax(&boxi[b*4+3], x);
  }
}

__global__ void k_boxfinal(const int* __restrict__ boxi, float* __restrict__ out)
{
  int b = threadIdx.x;
  if (b < BATCH) {
    int rmin = boxi[b*4+0], rmax = boxi[b*4+1];
    int cmin = boxi[b*4+2], cmax = boxi[b*4+3];
    bool has = rmax >= 0;
    out[OUT_BOXES + b*4 + 0] = has ? (float)cmin : 0.f;
    out[OUT_BOXES + b*4 + 1] = has ? (float)rmin : 0.f;
    out[OUT_BOXES + b*4 + 2] = has ? (float)cmax : 1023.f;
    out[OUT_BOXES + b*4 + 3] = has ? (float)rmax : 1023.f;
  }
}

__global__ void k_labels(float* out)
{
  int t = threadIdx.x;
  if (t < BATCH*NPTS) out[OUT_LABELS + t] = 1.0f;
}

// ---------------- peak picking: collect -> bitonic top-512 -> greedy NMS ---
__global__ void __launch_bounds__(256)
k_peaks(const float* __restrict__ sal, float* __restrict__ out_coords)
{
  __shared__ unsigned long long keys[CANDS];
  __shared__ unsigned long long red[256];
  __shared__ int s_cnt, s_nchosen, s_take;
  __shared__ int sel[NPTS], fb[NPTS];
  __shared__ unsigned char avail[KPEAKS];
  int b = blockIdx.x, tid = threadIdx.x;
  const float* s = sal + ((size_t)b << 18);
  for (int i = tid; i < CANDS; i += 256) keys[i] = 0ull;
  for (int i = tid; i < KPEAKS; i += 256) avail[i] = 1;
  if (tid == 0) { s_cnt = 0; s_nchosen = 0; }
  __syncthreads();
  // phase 1: collect 3x3 local maxima (ties count, sigmoid output > 0 always)
  for (int i = tid; i < HW; i += 256) {
    int y = i >> 9, x = i & 511;
    float v = s[i];
    float nb = -1e30f;
    for (int dy = -1; dy <= 1; ++dy)
      for (int dx = -1; dx <= 1; ++dx) {
        if (dy == 0 && dx == 0) continue;
        int yy = y+dy, xx = x+dx;
        if ((unsigned)yy < 512u && (unsigned)xx < 512u)
          nb = fmaxf(nb, s[(yy<<9)+xx]);
      }
    if (v >= nb && v > 0.f) {
      int slot = atomicAdd(&s_cnt, 1);
      if (slot < CANDS)
        keys[slot] = ((unsigned long long)__float_as_uint(v) << 32)
                   | (unsigned)(0xFFFFFFFFu - (unsigned)i);
    }
  }
  __syncthreads();
  // phase 2: bitonic sort descending (score desc, index asc on ties)
  for (int k = 2; k <= CANDS; k <<= 1) {
    for (int j = k >> 1; j > 0; j >>= 1) {
      for (int i = tid; i < CANDS; i += 256) {
        int ixj = i ^ j;
        if (ixj > i) {
          unsigned long long a = keys[i], c = keys[ixj];
          bool up = ((i & k) == 0);
          if (up ? (a < c) : (a > c)) { keys[i] = c; keys[ixj] = a; }
        }
      }
      __syncthreads();
    }
  }
  int ncand = min(s_cnt, KPEAKS);
  // phase 3: greedy spatial NMS, min_dist = 25 (dist^2 < 625)
  for (int i = 0; i < KPEAKS; ++i) {
    if (tid == 0) {
      bool valid = (i < ncand) && (keys[i] != 0ull);
      int take = (avail[i] && valid && s_nchosen < NPTS) ? 1 : 0;
      if (take) {
        unsigned idx = 0xFFFFFFFFu - (unsigned)(keys[i] & 0xFFFFFFFFull);
        sel[s_nchosen] = (int)idx;
        s_nchosen++;
      }
      s_take = take;
    }
    __syncthreads();
    if (s_take) {
      unsigned idx = 0xFFFFFFFFu - (unsigned)(keys[i] & 0xFFFFFFFFull);
      float xi = (float)(idx & 511u), yi = (float)(idx >> 9);
      for (int j = i + 1 + tid; j < KPEAKS; j += 256) {
        unsigned long long kj = keys[j];
        if (kj == 0ull) continue;
        unsigned jdx = 0xFFFFFFFFu - (unsigned)(kj & 0xFFFFFFFFull);
        float dx = xi - (float)(jdx & 511u);
        float dy = yi - (float)(jdx >> 9);
        if (dx*dx + dy*dy < 625.f) avail[j] = 0;
      }
    }
    __syncthreads();
  }
  // phase 4: fallback top-NPTS over non-peak saliency
  for (int q = 0; q < NPTS; ++q) {
    unsigned long long best = 0ull;
    for (int i = tid; i < HW; i += 256) {
      int y = i >> 9, x = i & 511;
      float v = s[i];
      float nb = -1e30f;
      for (int dy = -1; dy <= 1; ++dy)
        for (int dx = -1; dx <= 1; ++dx) {
          if (dy == 0 && dx == 0) continue;
          int yy = y+dy, xx = x+dx;
          if ((unsigned)yy < 512u && (unsigned)xx < 512u)
            nb = fmaxf(nb, s[(yy<<9)+xx]);
        }
      if (v >= nb && v > 0.f) continue;  // exclude peaks
      bool used = false;
      for (int r = 0; r < q; ++r) if (fb[r] == i) used = true;
      if (used) continue;
      unsigned long long key = ((unsigned long long)__float_as_uint(v) << 32)
                             | (unsigned)(0xFFFFFFFFu - (unsigned)i);
      if (key > best) best = key;
    }
    red[tid] = best;
    __syncthreads();
    for (int st = 128; st > 0; st >>= 1) {
      if (tid < st && red[tid+st] > red[tid]) red[tid] = red[tid+st];
      __syncthreads();
    }
    if (tid == 0) fb[q] = (int)(0xFFFFFFFFu - (unsigned)(red[0] & 0xFFFFFFFFull));
    __syncthreads();
  }
  // phase 5: write normalized coords
  if (tid < NPTS) {
    int n = s_nchosen;
    int d = tid - n;
    if (d < 0) d = 0;
    if (d > NPTS-1) d = NPTS-1;
    int idx = (tid < n) ? sel[tid] : fb[d];
    out_coords[b*NPTS*2 + tid*2 + 0] = (float)(idx & 511) * (1.f/512.f);
    out_coords[b*NPTS*2 + tid*2 + 1] = (float)(idx >> 9) * (1.f/512.f);
  }
}

// ---------------------------------------------------------------------------
extern "C" void kernel_launch(void* const* d_in, const int* in_sizes, int n_in,
                              void* d_out, int out_size, void* d_ws, size_t ws_size,
                              hipStream_t stream)
{
  const float* rgb   = (const float*)d_in[0];
  const float* dep   = (const float*)d_in[1];
  const float* w1c   = (const float*)d_in[2];
  const float* b1c   = (const float*)d_in[3];
  const float* gam   = (const float*)d_in[4];
  const float* bet   = (const float*)d_in[5];
  const float* mu    = (const float*)d_in[6];
  const float* var   = (const float*)d_in[7];
  const float* w2c   = (const float*)d_in[8];
  const float* b2c   = (const float*)d_in[9];
  float* out = (float*)d_out;

  char* ws = (char*)d_ws;          // needs ~59 MB
  h16*   fq   = (h16*)(ws + WS_FREQ);
  h16*   wB   = (h16*)(ws + WS_WB);
  float* sal  = (float*)(ws + WS_SAL);
  int*   boxi = (int*)(ws + WS_BOXI);
  float* masks = out + OUT_MASKS;

  k_wt      <<<16, 256, 0, stream>>>(w1c, wB);
  k_freq    <<<(BATCH*HW)/256, 256, 0, stream>>>(rgb, dep, fq);
  k_smooth  <<<(BATCH*2*HW)/256, 256, 0, stream>>>(fq);
  k_conv    <<<BATCH*2048, 256, 0, stream>>>(fq, wB, b1c, gam, bet, mu, var,
                                             w2c, b2c, sal);
  k_resize  <<<(BATCH*1048576)/256, 256, 0, stream>>>(sal, masks);
  k_boxinit <<<1, 64, 0, stream>>>(boxi);
  k_boxreduce<<<(BATCH*1048576)/256, 256, 0, stream>>>(masks, boxi);
  k_boxfinal<<<1, 32, 0, stream>>>(boxi, out);
  k_labels  <<<1, 128, 0, stream>>>(out);
  k_peaks   <<<BATCH, 256, 0, stream>>>(sal, out);
}
